// WindowAttention_67637144978019
// MI455X (gfx1250) — compile-verified, matrix operand order checked by reading, not run on hardware
//
#include <hip/hip_runtime.h>

// ---------------------------------------------------------------------------
// Window attention, fully fused, bf16 WMMA (v_wmma_f32_16x16x32_bf16) path.
// B_=8192 windows, N=64 tokens, C=256 channels, H=8 heads, hd=32.
// One block per window, 8 waves (wave32); one head per wave in attention.
// Tile loops deliberately NOT unrolled to stay under 256 VGPRs (avoid
// s_set_vgpr_msb thrash); only the K-chain is unrolled for the WMMA pipeline.
// ---------------------------------------------------------------------------

#define TOK   64
#define CH    256
#define HEADS 8
#define HD    32
#define NTHREADS 256

typedef __attribute__((ext_vector_type(16))) __bf16 v16bf;
typedef __attribute__((ext_vector_type(8)))  float  v8f;
typedef unsigned int       u32;
typedef unsigned short     u16;

union Frag {
    u32   u[8];
    v16bf v;
};

// f32 -> bf16 (round to nearest even)
__device__ inline u16 f2bf(float f) {
    u32 u = __builtin_bit_cast(u32, f);
    u += 0x7FFFu + ((u >> 16) & 1u);
    return (u16)(u >> 16);
}

// CDNA5 16-bit A/B fragment K map: VGPR v, lane-group g (lane>>4)
// V0..3 -> K = 2v + 8g ; V4..7 -> K = 16 + 2(v-4) + 8g  (each holds pair K,K+1)
__device__ inline int kmap(int v, int g) {
    return (v < 4) ? (2 * v + 8 * g) : (16 + 2 * (v - 4) + 8 * g);
}

// Load a 16x32 bf16 A-fragment (or B-fragment of a matrix stored so that each
// lane's K-pairs are contiguous) from LDS. base: row-major, `ld` elements/row.
__device__ inline v16bf frag_lds(const u16* base, int row0, int ld, int kbase, int lane) {
    int r = lane & 15, g = lane >> 4;
    const u16* p = base + (row0 + r) * ld + kbase;
    Frag f;
#pragma unroll
    for (int v = 0; v < 8; ++v)
        f.u[v] = *(const u32*)(p + kmap(v, g));
    return f.v;
}

// Load a B-fragment from row-major bf16 weights W[O][C]: column n of B == row
// (orow0+n) of W, contiguous over K.
__device__ inline v16bf frag_w(const u16* __restrict__ w, int orow0, int ldc, int kbase, int lane) {
    int r = lane & 15, g = lane >> 4;
    const u16* p = w + (orow0 + r) * ldc + kbase;
    Frag f;
#pragma unroll
    for (int v = 0; v < 8; ++v)
        f.u[v] = *(const u32*)(p + kmap(v, g));
    return f.v;
}

__device__ inline v8f wmma_bf16(v16bf a, v16bf b, v8f c) {
    return __builtin_amdgcn_wmma_f32_16x16x32_bf16(
        /*neg_a=*/false, a, /*neg_b=*/false, b,
        /*c_mod=*/(short)0, c, /*reuse_a=*/false, /*reuse_b=*/false);
}

// ---------------------------------------------------------------------------
// Kernel 0: convert f32 weights to bf16 in workspace.
// ---------------------------------------------------------------------------
__global__ __launch_bounds__(256) void wa_convert_weights(
    const float* __restrict__ qkv_w, const float* __restrict__ proj_w,
    u16* __restrict__ wq, u16* __restrict__ wp) {
    int i = blockIdx.x * blockDim.x + threadIdx.x;
    const int n1 = 3 * CH * CH;  // 196608
    const int n2 = CH * CH;      // 65536
    if (i < n1)            wq[i] = f2bf(qkv_w[i]);
    else if (i < n1 + n2)  wp[i - n1] = f2bf(proj_w[i - n1]);
}

// ---------------------------------------------------------------------------
// Kernel 1: fused window attention.
// ---------------------------------------------------------------------------
__global__ __launch_bounds__(NTHREADS) void wa_fused(
    const float* __restrict__ x,        // [B, 64, 256]
    const u16*   __restrict__ wqkv,     // [768, 256] bf16
    const float* __restrict__ bqkv,     // [768]
    const u16*   __restrict__ wproj,    // [256, 256] bf16
    const float* __restrict__ bproj,    // [256]
    const float* __restrict__ Ap,
    const float* __restrict__ Bp,
    float* __restrict__ out)            // [B, 64, 256]
{
    __shared__ u16   lds_x[TOK * CH];             // 32KB: x (bf16); reused as attn output
    __shared__ u16   lds_q[TOK * CH];             // 32KB: q * scale (bf16)
    __shared__ u16   lds_k[TOK * CH];             // 32KB: k (bf16)
    __shared__ u16   lds_vT[HEADS * HD * TOK];    // 32KB: v transposed per head [h][d][m]
    __shared__ u16   lds_p[HEADS * TOK * TOK];    // 64KB: softmax probs per head (bf16)
    __shared__ float lds_bias[TOK * TOK];         // 16KB: Laplacian bias table

    const int tid  = threadIdx.x;
    const int wave = tid >> 5;
    const int lane = tid & 31;
    const int g    = lane >> 4;
    const int l16  = lane & 15;
    const size_t bwin = blockIdx.x;

    // ---------------- Phase 0: stage x (packed pairs), build bias table ----
    {
        const float2* xb2 = (const float2*)(x + bwin * (TOK * CH));
        u32* lx = (u32*)lds_x;
        for (int i = tid; i < TOK * CH / 2; i += NTHREADS) {
            float2 xv = xb2[i];
            lx[i] = (u32)f2bf(xv.x) | ((u32)f2bf(xv.y) << 16);
        }
    }
    {
        float A  = Ap[0];
        float Bv = Bp[0];
        float A2 = A * A;
        float invB2 = 1.0f / (Bv * Bv);
        for (int i = tid; i < TOK * TOK; i += NTHREADS) {
            int ii = i >> 6, jj = i & 63;
            int d = abs((ii >> 3) - (jj >> 3)) + abs((ii & 7) - (jj & 7));
            lds_bias[i] = A2 * __expf(-0.5f * (float)d * invB2);
        }
    }
    __syncthreads();

    // ---------------- Phase 1: QKV GEMM (64x768 = x @ qkv_w^T) -------------
    // Three branch-free passes: Q tiles [0,16), K tiles [16,32), V tiles [32,48).
    const float qscale = 0.17677669529663689f;  // hd^-0.5

    // ---- Q: output columns [0,256), scaled ----
#pragma unroll 1
    for (int i = 0; i < 2; ++i) {
        int nt = wave + i * 8;            // 0..15
        int o0 = nt * 16;
        int ocol = o0 + l16;
        float bo = bqkv[ocol];
#pragma unroll 1
        for (int mt = 0; mt < 4; ++mt) {
            v8f acc = {};
#pragma unroll
            for (int kt = 0; kt < 8; ++kt) {
                v16bf a = frag_lds(lds_x, mt * 16, CH, kt * 32, lane);
                v16bf b = frag_w(wqkv, o0, CH, kt * 32, lane);
                acc = wmma_bf16(a, b, acc);
            }
#pragma unroll
            for (int v = 0; v < 8; ++v) {
                int m = mt * 16 + v + 8 * g;
                lds_q[m * CH + ocol] = f2bf((acc[v] + bo) * qscale);
            }
        }
    }

    // ---- K: output columns [256,512) ----
#pragma unroll 1
    for (int i = 0; i < 2; ++i) {
        int nt = 16 + wave + i * 8;       // 16..31
        int o0 = nt * 16;
        int ocol = o0 + l16;
        float bo = bqkv[ocol];
        int kcol = ocol - 256;
#pragma unroll 1
        for (int mt = 0; mt < 4; ++mt) {
            v8f acc = {};
#pragma unroll
            for (int kt = 0; kt < 8; ++kt) {
                v16bf a = frag_lds(lds_x, mt * 16, CH, kt * 32, lane);
                v16bf b = frag_w(wqkv, o0, CH, kt * 32, lane);
                acc = wmma_bf16(a, b, acc);
            }
#pragma unroll
            for (int v = 0; v < 8; ++v) {
                int m = mt * 16 + v + 8 * g;
                lds_k[m * CH + kcol] = f2bf(acc[v] + bo);
            }
        }
    }

    // ---- V: output columns [512,768), stored transposed per head ----
#pragma unroll 1
    for (int i = 0; i < 2; ++i) {
        int nt = 32 + wave + i * 8;       // 32..47
        int o0 = nt * 16;
        int ocol = o0 + l16;
        float bo = bqkv[ocol];
        int oo = ocol - 512;              // 0..255; head = oo>>5, d = oo&31
        int vbase = (oo >> 5) * (HD * TOK) + (oo & 31) * TOK;  // lds_vT[h][d][.]
#pragma unroll 1
        for (int mt = 0; mt < 4; ++mt) {
            v8f acc = {};
#pragma unroll
            for (int kt = 0; kt < 8; ++kt) {
                v16bf a = frag_lds(lds_x, mt * 16, CH, kt * 32, lane);
                v16bf b = frag_w(wqkv, o0, CH, kt * 32, lane);
                acc = wmma_bf16(a, b, acc);
            }
#pragma unroll
            for (int v = 0; v < 8; ++v) {
                int m = mt * 16 + v + 8 * g;
                lds_vT[vbase + m] = f2bf(acc[v] + bo);
            }
        }
    }
    __syncthreads();

    // ---------------- Phase 2: attention, one head per wave ----------------
    {
        const int h = wave;
        const u16* qh = lds_q + h * HD;              // row stride CH
        const u16* kh = lds_k + h * HD;              // row stride CH
        const u16* vT = lds_vT + h * HD * TOK;       // [d][m], row stride TOK
        u16* ph = lds_p + h * TOK * TOK;             // [m][n], row stride TOK

        // scores + softmax, one 16-row block (mt) at a time
#pragma unroll 1
        for (int mt = 0; mt < 4; ++mt) {
            v16bf a = frag_lds(qh, mt * 16, CH, 0, lane);  // 16x32, K = hd
            v8f s[4];
#pragma unroll
            for (int nt = 0; nt < 4; ++nt) {
                // B col n = row (nt*16+n) of k_h, contiguous over d
                v16bf bk = frag_lds(kh, nt * 16, CH, 0, lane);
                v8f z = {};
                s[nt] = wmma_bf16(a, bk, z);
#pragma unroll
                for (int v = 0; v < 8; ++v) {
                    int m = mt * 16 + v + 8 * g;
                    int n = nt * 16 + l16;
                    s[nt][v] += lds_bias[m * TOK + n];
                }
            }
            // row-wise softmax: 64 values per row live in 4 tiles x 16 lanes
#pragma unroll
            for (int v = 0; v < 8; ++v) {
                float mx = fmaxf(fmaxf(s[0][v], s[1][v]), fmaxf(s[2][v], s[3][v]));
#pragma unroll
                for (int off = 8; off >= 1; off >>= 1)
                    mx = fmaxf(mx, __shfl_xor(mx, off, 32));
                float e[4];
                float sum = 0.0f;
#pragma unroll
                for (int nt = 0; nt < 4; ++nt) {
                    e[nt] = __expf(s[nt][v] - mx);
                    sum += e[nt];
                }
#pragma unroll
                for (int off = 8; off >= 1; off >>= 1)
                    sum += __shfl_xor(sum, off, 32);
                float r = 1.0f / sum;
                int m = mt * 16 + v + 8 * g;
#pragma unroll
                for (int nt = 0; nt < 4; ++nt)
                    ph[m * TOK + nt * 16 + l16] = f2bf(e[nt] * r);
            }
        }

        // out_h = P(64x64) @ v_h(64x32); B col n = row (dt*16+n) of vT over m'
#pragma unroll 1
        for (int mt = 0; mt < 4; ++mt) {
#pragma unroll
            for (int dt = 0; dt < 2; ++dt) {
                v8f acc = {};
#pragma unroll
                for (int kt = 0; kt < 2; ++kt) {
                    v16bf a  = frag_lds(ph, mt * 16, TOK, kt * 32, lane);
                    v16bf bv = frag_lds(vT, dt * 16, TOK, kt * 32, lane);
                    acc = wmma_bf16(a, bv, acc);
                }
#pragma unroll
                for (int v = 0; v < 8; ++v) {
                    int m = mt * 16 + v + 8 * g;
                    int d = dt * 16 + l16;
                    lds_x[m * CH + h * HD + d] = f2bf(acc[v]);  // reuse lds_x as attn out
                }
            }
        }
    }
    __syncthreads();

    // ---------------- Phase 3: output projection ---------------------------
    float* ob = out + bwin * (TOK * CH);
#pragma unroll 1
    for (int i = 0; i < 2; ++i) {
        int nt = wave + i * 8;            // 0..15
        int o0 = nt * 16;
        int ocol = o0 + l16;
        float bo = bproj[ocol];
#pragma unroll 1
        for (int mt = 0; mt < 4; ++mt) {
            v8f acc = {};
#pragma unroll
            for (int kt = 0; kt < 8; ++kt) {
                v16bf a = frag_lds(lds_x, mt * 16, CH, kt * 32, lane);
                v16bf b = frag_w(wproj, o0, CH, kt * 32, lane);
                acc = wmma_bf16(a, b, acc);
            }
#pragma unroll
            for (int v = 0; v < 8; ++v) {
                int m = mt * 16 + v + 8 * g;
                ob[m * CH + ocol] = acc[v] + bo;
            }
        }
    }
}

// ---------------------------------------------------------------------------
extern "C" void kernel_launch(void* const* d_in, const int* in_sizes, int n_in,
                              void* d_out, int out_size, void* d_ws, size_t ws_size,
                              hipStream_t stream) {
    const float* x      = (const float*)d_in[0];
    const float* qkv_w  = (const float*)d_in[1];
    const float* qkv_b  = (const float*)d_in[2];
    const float* proj_w = (const float*)d_in[3];
    const float* proj_b = (const float*)d_in[4];
    const float* A      = (const float*)d_in[5];
    const float* B      = (const float*)d_in[6];
    float* out = (float*)d_out;

    u16* wq = (u16*)d_ws;                 // 768*256 bf16
    u16* wp = wq + 3 * CH * CH;           // 256*256 bf16

    int nwin = in_sizes[0] / (TOK * CH);  // 8192

    int nconv = 3 * CH * CH + CH * CH;
    wa_convert_weights<<<(nconv + 255) / 256, 256, 0, stream>>>(qkv_w, proj_w, wq, wp);
    wa_fused<<<nwin, NTHREADS, 0, stream>>>(x, wq, qkv_b, wp, proj_b, A, B, out);
}
